// MultiHeadAttentionBlock_38053410242756
// MI455X (gfx1250) — matrix operand order repaired and checked by reading, not compiled, not run on hardware
//
#include <hip/hip_runtime.h>

// ---------------------------------------------------------------------------
// MHA block for gfx1250 (MI455X), wave32, WMMA f16 -> f32 accumulate.
//   B=4, S=2048, HID=1024, HEADS=16, Dh=64
// Async global->LDS staging (ASYNCcnt) + double-buffered LDS pipelines.
// ---------------------------------------------------------------------------

typedef __attribute__((ext_vector_type(16))) _Float16 v16h;
typedef __attribute__((ext_vector_type(8)))  float    v8f;

#define B_     4
#define S_     2048
#define HID_   1024
#define HEADS_ 16
#define DH_    64

#define WMMA_F16(a, b, c) \
  __builtin_amdgcn_wmma_f32_16x16x32_f16(false, (a), false, (b), (short)0, (c), false, false)

union FragH {
  v16h v;
  unsigned u[8];
  _Float16 h[16];
};

// CDNA5 async copy: 16B global -> LDS, tracked by ASYNCcnt.
__device__ inline void async_copy_b128(const void* gsrc, unsigned lds_off) {
  const unsigned long long ga = (unsigned long long)gsrc;
  asm volatile("global_load_async_to_lds_b128 %0, %1, off"
               :: "v"(lds_off), "v"(ga) : "memory");
}
__device__ inline void wait_async0() {
  asm volatile("s_wait_asynccnt 0x0" ::: "memory");
}
__device__ inline unsigned lds_addr_of(const void* p) {
  return (unsigned)(size_t)p;   // generic LDS addr: low 32 bits = LDS offset
}

// A-fragment (16x32 f16, M x K). p points at (m=0,k=0); row-major, stride ld halves.
__device__ inline v16h load_a_frag(const _Float16* p, int ld, int k0) {
  const int lane = threadIdx.x & 31;
  const int m    = lane & 15;
  const int hi   = lane >> 4;
  const _Float16* row = p + (size_t)m * ld + k0;
  FragH f;
#pragma unroll
  for (int v = 0; v < 8; ++v) {
    const int k = (v << 1) + (hi << 4);
    f.u[v] = *reinterpret_cast<const unsigned*>(row + k);
  }
  return f.v;
}

// B-fragment (32x16 f16, K x N) from an [n][k] row-major buffer (B[k][n]=src[n][k]).
__device__ inline v16h load_b_frag_nk(const _Float16* p, int ld, int n0, int k0) {
  const int lane = threadIdx.x & 31;
  const int n    = lane & 15;
  const int kb   = (lane >> 4) << 4;
  const _Float16* row = p + (size_t)(n0 + n) * ld + k0 + kb;
  FragH f;
#pragma unroll
  for (int v = 0; v < 8; ++v)
    f.u[v] = *reinterpret_cast<const unsigned*>(row + 2 * v);
  return f.v;
}

__device__ inline v8f zero8() {
  v8f z = {0.f, 0.f, 0.f, 0.f, 0.f, 0.f, 0.f, 0.f};
  return z;
}

// ---------------------------------------------------------------------------
// f32 -> f16 conversion (weights)
// ---------------------------------------------------------------------------
__global__ void f32_to_f16_kernel(const float* __restrict__ src,
                                  _Float16* __restrict__ dst, int n) {
  int i = blockIdx.x * blockDim.x + threadIdx.x;
  const int stride = gridDim.x * blockDim.x;
  for (; i < n; i += stride) dst[i] = (_Float16)src[i];
}

// ---------------------------------------------------------------------------
// Projection GEMM: C[m,n] = A[m,:] . W[n,:] + bias[n]   (A: MxK, W: NxK f16)
// WG tile 128(M) x 128(N), 256 threads = 8 waves (4 m-sub x 2 n-sub),
// wave tile 32x64, k-step 32, double-buffered LDS, async W (and async f16 A).
// MODE 0: f16 output, [b][h][s][dh] head-split layout (Q/K/V).
// MODE 1: f32 output, row-major [m][HID] (final projection into d_out).
// ---------------------------------------------------------------------------
template <typename AT, int MODE>
__global__ __launch_bounds__(256) void gemm_wmma_kernel(
    const AT* __restrict__ A, const _Float16* __restrict__ W,
    const float* __restrict__ bias, void* __restrict__ Out, int K) {
  constexpr int LDT = 40;  // 32 + 8 halves pad
  __shared__ alignas(16) _Float16 As[2][128 * LDT];
  __shared__ alignas(16) _Float16 Ws[2][128 * LDT];

  const int t    = threadIdx.x;
  const int lane = t & 31;
  const int w    = t >> 5;
  const int mw   = w & 3;        // m sub-tile: mw*32
  const int nw   = w >> 2;       // n sub-tile: nw*64
  const int n0blk = blockIdx.x * 128;
  const int m0blk = blockIdx.y * 128;
  const int NK    = K / 32;

  float4 areg[4];  // f32-A pipeline registers

  auto issueW = [&](int kt, int bufi) {
#pragma unroll
    for (int j = 0; j < 2; ++j) {
      const int idx = t * 2 + j, row = idx >> 2, ch = idx & 3;  // 128 rows x 4 chunks
      async_copy_b128(W + (size_t)(n0blk + row) * K + kt + ch * 8,
                      lds_addr_of(&Ws[bufi][row * LDT + ch * 8]));
    }
  };
  auto loadA = [&](int kt, int bufi) {
    if (sizeof(AT) == 4) {
      const float* Af = reinterpret_cast<const float*>(A);
#pragma unroll
      for (int j = 0; j < 4; ++j) {
        const int idx = t * 4 + j, row = idx >> 3, ch = idx & 7;  // 128 x 8 float4
        areg[j] = *reinterpret_cast<const float4*>(
            Af + (size_t)(m0blk + row) * K + kt + ch * 4);
      }
    } else {
      const _Float16* Ah = reinterpret_cast<const _Float16*>(A);
#pragma unroll
      for (int j = 0; j < 2; ++j) {
        const int idx = t * 2 + j, row = idx >> 2, ch = idx & 3;  // 128 x 4 uint4
        async_copy_b128(Ah + (size_t)(m0blk + row) * K + kt + ch * 8,
                        lds_addr_of(&As[bufi][row * LDT + ch * 8]));
      }
    }
  };
  auto storeA = [&](int bufi) {
    if (sizeof(AT) == 4) {
#pragma unroll
      for (int j = 0; j < 4; ++j) {
        const int idx = t * 4 + j, row = idx >> 3, ch = idx & 7;
        union { _Float16 h[4]; uint2 u; } pk;
        pk.h[0] = (_Float16)areg[j].x; pk.h[1] = (_Float16)areg[j].y;
        pk.h[2] = (_Float16)areg[j].z; pk.h[3] = (_Float16)areg[j].w;
        *reinterpret_cast<uint2*>(&As[bufi][row * LDT + ch * 4]) = pk.u;
      }
    }
  };

  v8f acc[2][4];
#pragma unroll
  for (int h = 0; h < 2; ++h)
#pragma unroll
    for (int i = 0; i < 4; ++i) acc[h][i] = zero8();

  // prologue: stage k-tile 0 into buffer 0
  loadA(0, 0);
  issueW(0, 0);
  storeA(0);
  wait_async0();
  __syncthreads();

  for (int kts = 0; kts < NK; ++kts) {
    const int cur = kts & 1, nxt = cur ^ 1;
    const bool more = (kts + 1 < NK);
    if (more) {                       // prefetch next tile behind the WMMAs
      loadA((kts + 1) * 32, nxt);
      issueW((kts + 1) * 32, nxt);
    }

    const _Float16* Ab = &As[cur][(mw * 32) * LDT];
    const v16h a0 = load_a_frag(Ab, LDT, 0);
    const v16h a1 = load_a_frag(Ab + 16 * LDT, LDT, 0);
#pragma unroll
    for (int nf = 0; nf < 4; ++nf) {
      const v16h b = load_b_frag_nk(&Ws[cur][0], LDT, nw * 64 + nf * 16, 0);
      acc[0][nf] = WMMA_F16(a0, b, acc[0][nf]);
      acc[1][nf] = WMMA_F16(a1, b, acc[1][nf]);
    }

    if (more) {
      storeA(nxt);
      wait_async0();
    }
    __syncthreads();
  }

  // ---- epilogue ----
  const int hi = lane >> 4, n = lane & 15;
#pragma unroll
  for (int nf = 0; nf < 4; ++nf) {
    const int ng = n0blk + nw * 64 + nf * 16 + n;
    const float bv = bias[ng];
#pragma unroll
    for (int half = 0; half < 2; ++half)
#pragma unroll
      for (int r = 0; r < 8; ++r) {
        const int mg = m0blk + mw * 32 + half * 16 + r + hi * 8;
        const float val = acc[half][nf][r] + bv;
        if (MODE == 0) {
          const int bb = mg >> 11, ss = mg & 2047;   // S_ = 2048
          const int hh = ng >> 6,  dd = ng & 63;     // DH_ = 64
          reinterpret_cast<_Float16*>(Out)[
              (((size_t)(bb * HEADS_ + hh) * S_ + ss) << 6) + dd] = (_Float16)val;
        } else {
          reinterpret_cast<float*>(Out)[(size_t)mg * HID_ + ng] = val;
        }
      }
  }
}

// ---------------------------------------------------------------------------
// Flash attention: one WG per (b, h, 64-row q tile). 128 threads = 4 waves,
// each wave owns 16 q rows. K tiles arrive via async global->LDS; V tiles are
// pipelined through registers and stored transposed. Double-buffered K/V,
// online softmax in the C/D fragment layout, P -> LDS -> A-fragment for PV.
// ---------------------------------------------------------------------------
__global__ __launch_bounds__(128) void flash_attn_kernel(
    const _Float16* __restrict__ Q, const _Float16* __restrict__ Km,
    const _Float16* __restrict__ Vm, _Float16* __restrict__ Outp) {
  constexpr int LDK = 72;  // 64 + 8 halves pad
  __shared__ alignas(16) _Float16 Ks[2][64 * LDK];     // [s][d]
  __shared__ alignas(16) _Float16 Vt[2][64 * LDK];     // [d][s] transposed
  __shared__ alignas(16) _Float16 Ps[4][16 * LDK];     // per-wave P tile [m][s]

  const int t = threadIdx.x, lane = t & 31, w = t >> 5;
  const int hi = lane >> 4, n = lane & 15;
  const int qt = blockIdx.x, h = blockIdx.y, b = blockIdx.z;
  const size_t bh = (size_t)(b * HEADS_ + h);

  // Q fragments (1/sqrt(Dh) folded in)
  const _Float16* Qbase = Q + (bh * S_ + (size_t)qt * 64 + (size_t)w * 16) * DH_;
  v16h qa0 = load_a_frag(Qbase, DH_, 0);
  v16h qa1 = load_a_frag(Qbase, DH_, 32);
#pragma unroll
  for (int i = 0; i < 16; ++i) {
    qa0[i] = (_Float16)((float)qa0[i] * 0.125f);
    qa1[i] = (_Float16)((float)qa1[i] * 0.125f);
  }

  uint4 vreg[4];  // V pipeline registers
  auto issueK = [&](int kt, int bufi) {
    const _Float16* Ksrc = Km + (bh * S_ + kt) * DH_;
#pragma unroll
    for (int j = 0; j < 4; ++j) {
      const int idx = t * 4 + j, row = idx >> 3, ch = idx & 7;  // 64 x 8 chunks
      async_copy_b128(Ksrc + row * DH_ + ch * 8,
                      lds_addr_of(&Ks[bufi][row * LDK + ch * 8]));
    }
  };
  auto loadV = [&](int kt) {
    const _Float16* Vsrc = Vm + (bh * S_ + kt) * DH_;
#pragma unroll
    for (int j = 0; j < 4; ++j) {
      const int idx = t * 4 + j, s = idx >> 3, ch = idx & 7;
      vreg[j] = *reinterpret_cast<const uint4*>(Vsrc + s * DH_ + ch * 8);
    }
  };
  auto storeVt = [&](int bufi) {
#pragma unroll
    for (int j = 0; j < 4; ++j) {
      const int idx = t * 4 + j, s = idx >> 3, ch = idx & 7;
      union { uint4 q; _Float16 hh[8]; } u;
      u.q = vreg[j];
#pragma unroll
      for (int i = 0; i < 8; ++i) Vt[bufi][(ch * 8 + i) * LDK + s] = u.hh[i];
    }
  };

  v8f acc[4];
#pragma unroll
  for (int i = 0; i < 4; ++i) acc[i] = zero8();
  float mrow[8], lrow[8];
#pragma unroll
  for (int r = 0; r < 8; ++r) { mrow[r] = -3.0e38f; lrow[r] = 0.f; }

  // prologue
  issueK(0, 0);
  loadV(0);
  storeVt(0);
  wait_async0();
  __syncthreads();

  constexpr int NT = S_ / 64;
  for (int it = 0; it < NT; ++it) {
    const int cur = it & 1, nxt = cur ^ 1;
    const bool more = (it + 1 < NT);
    if (more) {
      issueK((it + 1) * 64, nxt);
      loadV((it + 1) * 64);
    }

    // ---- scores: S = Q . K^T (16 x 64) ----
    v8f sc[4];
#pragma unroll
    for (int nf = 0; nf < 4; ++nf) {
      v8f z = zero8();
      z = WMMA_F16(qa0, load_b_frag_nk(&Ks[cur][0], LDK, nf * 16, 0),  z);
      z = WMMA_F16(qa1, load_b_frag_nk(&Ks[cur][0], LDK, nf * 16, 32), z);
      sc[nf] = z;
    }

    // ---- online softmax (row = r + 8*hi, dup across 16-lane group) ----
#pragma unroll
    for (int r = 0; r < 8; ++r) {
      float mx = fmaxf(fmaxf(sc[0][r], sc[1][r]), fmaxf(sc[2][r], sc[3][r]));
#pragma unroll
      for (int off = 1; off < 16; off <<= 1) mx = fmaxf(mx, __shfl_xor(mx, off, 32));
      const float mn = fmaxf(mrow[r], mx);
      const float corr = __expf(mrow[r] - mn);
      mrow[r] = mn;
      float rs = 0.f;
#pragma unroll
      for (int nf = 0; nf < 4; ++nf) {
        const float p = __expf(sc[nf][r] - mn);
        sc[nf][r] = p;
        rs += p;
      }
#pragma unroll
      for (int off = 1; off < 16; off <<= 1) rs += __shfl_xor(rs, off, 32);
      lrow[r] = lrow[r] * corr + rs;
#pragma unroll
      for (int nf = 0; nf < 4; ++nf) acc[nf][r] *= corr;
    }

    // ---- P -> LDS (f16), then PV WMMA ----
    _Float16* Pw = Ps[w];
#pragma unroll
    for (int nf = 0; nf < 4; ++nf)
#pragma unroll
      for (int r = 0; r < 8; ++r)
        Pw[(r + hi * 8) * LDK + nf * 16 + n] = (_Float16)sc[nf][r];
    const v16h pa0 = load_a_frag(Pw, LDK, 0);
    const v16h pa1 = load_a_frag(Pw, LDK, 32);
#pragma unroll
    for (int nf = 0; nf < 4; ++nf) {
      acc[nf] = WMMA_F16(pa0, load_b_frag_nk(&Vt[cur][0], LDK, nf * 16, 0),  acc[nf]);
      acc[nf] = WMMA_F16(pa1, load_b_frag_nk(&Vt[cur][0], LDK, nf * 16, 32), acc[nf]);
    }

    if (more) {
      storeVt(nxt);
      wait_async0();
    }
    __syncthreads();
  }

  // ---- normalize + write merged-head f16 output [b][s][h*64+d] ----
  const int srow = qt * 64 + w * 16;
#pragma unroll
  for (int r = 0; r < 8; ++r) {
    const float inv = 1.0f / lrow[r];
    const int sg = srow + r + hi * 8;
    const size_t base = ((size_t)b * S_ + sg) * HID_ + (size_t)h * DH_;
#pragma unroll
    for (int nf = 0; nf < 4; ++nf)
      Outp[base + nf * 16 + n] = (_Float16)(acc[nf][r] * inv);
  }
}

// ---------------------------------------------------------------------------
// Host-side launch
// ---------------------------------------------------------------------------
extern "C" void kernel_launch(void* const* d_in, const int* in_sizes, int n_in,
                              void* d_out, int out_size, void* d_ws, size_t ws_size,
                              hipStream_t stream) {
  (void)in_sizes; (void)n_in; (void)out_size; (void)ws_size;
  const float* x1 = (const float*)d_in[0];
  const float* x2 = (const float*)d_in[1];
  const float* Wq = (const float*)d_in[2];
  const float* bq = (const float*)d_in[3];
  const float* Wk = (const float*)d_in[4];
  const float* bk = (const float*)d_in[5];
  const float* Wv = (const float*)d_in[6];
  const float* bv = (const float*)d_in[7];
  const float* Wo = (const float*)d_in[8];
  const float* bo = (const float*)d_in[9];

  const size_t WN = (size_t)HID_ * HID_;          // 1 Mi elems
  const size_t XN = (size_t)B_ * S_ * HID_;       // 8 Mi elems
  _Float16* Wqh = (_Float16*)d_ws;
  _Float16* Wkh = Wqh + WN;
  _Float16* Wvh = Wkh + WN;
  _Float16* Woh = Wvh + WN;
  _Float16* Qb  = Woh + WN;
  _Float16* Kb  = Qb + XN;
  _Float16* Vb  = Kb + XN;
  _Float16* Ao  = Vb + XN;   // total ws use ~75 MB

  f32_to_f16_kernel<<<1024, 256, 0, stream>>>(Wq, Wqh, (int)WN);
  f32_to_f16_kernel<<<1024, 256, 0, stream>>>(Wk, Wkh, (int)WN);
  f32_to_f16_kernel<<<1024, 256, 0, stream>>>(Wv, Wvh, (int)WN);
  f32_to_f16_kernel<<<1024, 256, 0, stream>>>(Wo, Woh, (int)WN);

  const dim3 gblk(HID_ / 128, (B_ * S_) / 128);   // 8 x 64 WGs
  gemm_wmma_kernel<float, 0><<<gblk, 256, 0, stream>>>(x1, Wqh, bq, Qb, HID_);
  gemm_wmma_kernel<float, 0><<<gblk, 256, 0, stream>>>(x2, Wkh, bk, Kb, HID_);
  gemm_wmma_kernel<float, 0><<<gblk, 256, 0, stream>>>(x2, Wvh, bv, Vb, HID_);

  flash_attn_kernel<<<dim3(S_ / 64, HEADS_, B_), 128, 0, stream>>>(Qb, Kb, Vb, Ao);

  gemm_wmma_kernel<_Float16, 1><<<gblk, 256, 0, stream>>>(Ao, Woh, bo, d_out, HID_);
}
